// GroupASL_53197464928603
// MI455X (gfx1250) — matrix operand order repaired and checked by reading, not compiled, not run on hardware
//
#include <hip/hip_runtime.h>
#include <hip/hip_bf16.h>

// GroupASL: out[b,g,k] = <h[b,g,:]/||.||, w[g,:,k]/||.||>
// Fused as: raw batched GEMM (split-bf16 fp32 emulation on WMMA) with
// row/col squared-norm reductions folded into the tile staging loads,
// inverse-norm scaling applied in the epilogue.

typedef __attribute__((ext_vector_type(16))) __bf16 v16bf;
typedef __attribute__((ext_vector_type(8)))  float  v8f;

#define B_SZ 512
#define G_SZ 100
#define D_SZ 768
#define K_SZ 100

#define BM 128        // rows (b) per block
#define BN 112        // 7 n-tiles of 16, covers K=100 with masking
#define BK 32         // reduction step (bf16 WMMA K)
#define LDA 40        // padded LDS stride (halfs): keeps 16B alignment, spreads banks
#define LDB 40
#define NT  7         // 16-wide n-tiles per wave

union Frag32B { v16bf v; uint4 q[2]; };

__device__ __forceinline__ void split2(float x, unsigned short& hi, unsigned short& lo) {
  // hi = truncate-to-bf16(x); lo = truncate-to-bf16(x - hi). hi*hi exact in f32;
  // hi*lo terms recover the next 8 mantissa bits -> ~2^-16 relative error.
  unsigned u  = __float_as_uint(x);
  unsigned short h = (unsigned short)(u >> 16);
  float hf = __uint_as_float(((unsigned)h) << 16);
  float r  = x - hf;
  lo = (unsigned short)(__float_as_uint(r) >> 16);
  hi = h;
}

__global__ __launch_bounds__(256)
__attribute__((amdgpu_num_vgpr(256)))       // stay under the VGPR-MSB boundary:
                                            // no s_set_vgpr_msb churn, better occupancy
void groupasl_gemm(
    const float* __restrict__ h,
    const float* __restrict__ w,
    float* __restrict__ out)
{
  __shared__ __align__(16) unsigned short sAhi[BM * LDA];
  __shared__ __align__(16) unsigned short sAlo[BM * LDA];
  __shared__ __align__(16) unsigned short sBhi[BN * LDB];
  __shared__ __align__(16) unsigned short sBlo[BN * LDB];
  __shared__ float sRowSq[BM];
  __shared__ float sColSq[BN];

  const int t     = threadIdx.x;
  const int lane  = t & 31;
  const int wv    = t >> 5;        // wave id 0..7 -> owns 16 M rows
  const int l16   = lane & 15;
  const int lhalf = lane >> 4;     // 0: lanes 0-15, 1: lanes 16-31

  const int g     = blockIdx.y;
  const int bbase = blockIdx.x * BM;

  const float* __restrict__ wg = w + (size_t)g * (D_SZ * K_SZ);

  if (t < BM) sRowSq[t] = 0.0f;
  if (t < BN) sColSq[t] = 0.0f;

  float rowSq[4] = {0.f, 0.f, 0.f, 0.f};
  // k index of B element i repeats with period 7 (256*7 % 112 == 0),
  // so 7 accumulators suffice: colSq[i % 7] always sees the same k.
  float colSq[7];
#pragma unroll
  for (int i = 0; i < 7; ++i) colSq[i] = 0.0f;

  v8f acc[NT] = {};

#pragma unroll 1
  for (int kk = 0; kk < D_SZ; kk += BK) {
    // ---- stage A tile: BM x BK f32 -> bf16 hi/lo, fuse row sum-of-squares ----
    // 128 rows * 8 float4 = 1024 float4 loads / 256 threads = 4 each
#pragma unroll
    for (int i = 0; i < 4; ++i) {
      const int f   = t + i * 256;
      const int row = f >> 3;       // 8 float4 per row
      const int c4  = f & 7;
      const float4 v = *(const float4*)(h +
          ((size_t)(bbase + row) * G_SZ + g) * D_SZ + kk + c4 * 4);
      rowSq[i] += v.x * v.x + v.y * v.y + v.z * v.z + v.w * v.w;
      unsigned short h0, h1, h2, h3, l0, l1, l2, l3;
      split2(v.x, h0, l0); split2(v.y, h1, l1);
      split2(v.z, h2, l2); split2(v.w, h3, l3);
      uint2 ph = make_uint2((unsigned)h0 | ((unsigned)h1 << 16),
                            (unsigned)h2 | ((unsigned)h3 << 16));
      uint2 pl = make_uint2((unsigned)l0 | ((unsigned)l1 << 16),
                            (unsigned)l2 | ((unsigned)l3 << 16));
      *(uint2*)&sAhi[row * LDA + c4 * 4] = ph;
      *(uint2*)&sAlo[row * LDA + c4 * 4] = pl;
    }

    // ---- stage B tile: BK x BN, stored column-major (k-major) in LDS, fuse col norms ----
    // 32*112 = 3584 scalar loads / 256 threads = 14 each; consecutive t -> consecutive k
#pragma unroll
    for (int i = 0; i < 14; ++i) {
      const int e = t + i * 256;
      const int d = e / BN;
      const int k = e - d * BN;
      const float v = (k < K_SZ) ? wg[(size_t)(kk + d) * K_SZ + k] : 0.0f;
      colSq[i % 7] += v * v;
      unsigned short hv, lv;
      split2(v, hv, lv);
      sBhi[k * LDB + d] = hv;
      sBlo[k * LDB + d] = lv;
    }
    __syncthreads();

    // ---- WMMA: per wave, 1 m-tile x 7 n-tiles, 3 bf16 WMMAs each (hi*hi + hi*lo + lo*hi)
    // A 16x32 bf16 layout: lanes 0-15 hold K chunks {0..7, 16..23}, lanes 16-31 {8..15, 24..31}
    Frag32B aHi, aLo;
    {
      const int row = wv * 16 + l16;
      const int o0  = row * LDA + lhalf * 8;   // halfs; *2 bytes -> 16B aligned
      aHi.q[0] = *(const uint4*)&sAhi[o0];
      aHi.q[1] = *(const uint4*)&sAhi[o0 + 16];
      aLo.q[0] = *(const uint4*)&sAlo[o0];
      aLo.q[1] = *(const uint4*)&sAlo[o0 + 16];
    }
#pragma unroll
    for (int j = 0; j < NT; ++j) {
      // B 32x16 bf16 layout: lane holds column n = l16, K chunks {lhalf*8..+7, 16+lhalf*8..+7}
      const int n  = j * 16 + l16;
      const int ob = n * LDB + lhalf * 8;
      Frag32B bHi, bLo;
      bHi.q[0] = *(const uint4*)&sBhi[ob];
      bHi.q[1] = *(const uint4*)&sBhi[ob + 16];
      bLo.q[0] = *(const uint4*)&sBlo[ob];
      bLo.q[1] = *(const uint4*)&sBlo[ob + 16];
      acc[j] = __builtin_amdgcn_wmma_f32_16x16x32_bf16(
          false, aHi.v, false, bHi.v, (short)0, acc[j], false, false);
      acc[j] = __builtin_amdgcn_wmma_f32_16x16x32_bf16(
          false, aHi.v, false, bLo.v, (short)0, acc[j], false, false);
      acc[j] = __builtin_amdgcn_wmma_f32_16x16x32_bf16(
          false, aLo.v, false, bHi.v, (short)0, acc[j], false, false);
    }
    __syncthreads();
  }

  // ---- reduce fused norm partial sums into LDS (ds_add_f32) ----
#pragma unroll
  for (int i = 0; i < 4; ++i)
    atomicAdd(&sRowSq[(t >> 3) + i * 32], rowSq[i]);
#pragma unroll
  for (int i = 0; i < 7; ++i) {
    const int e = t + i * 256;
    const int d = e / BN;
    const int k = e - d * BN;
    atomicAdd(&sColSq[k], colSq[i]);
  }
  __syncthreads();

  // ---- epilogue: scale by inverse norms, masked store ----
  const float EPS = 1e-12f;
  float sh[8];
#pragma unroll
  for (int r = 0; r < 8; ++r) {
    const int m = wv * 16 + lhalf * 8 + r;   // C layout: VGPR r -> M = r + 8*lhalf
    sh[r] = 1.0f / fmaxf(sqrtf(sRowSq[m]), EPS);
  }
#pragma unroll
  for (int j = 0; j < NT; ++j) {
    const int k = j * 16 + l16;              // C layout: N = lane%16
    if (k >= K_SZ) continue;
    const float sw = 1.0f / fmaxf(sqrtf(sColSq[k]), EPS);
#pragma unroll
    for (int r = 0; r < 8; ++r) {
      const int b = bbase + wv * 16 + lhalf * 8 + r;
      out[(size_t)b * (G_SZ * K_SZ) + g * K_SZ + k] = acc[j][r] * sh[r] * sw;
    }
  }
}

extern "C" void kernel_launch(void* const* d_in, const int* in_sizes, int n_in,
                              void* d_out, int out_size, void* d_ws, size_t ws_size,
                              hipStream_t stream) {
  (void)in_sizes; (void)n_in; (void)out_size; (void)d_ws; (void)ws_size;
  const float* h = (const float*)d_in[0];
  const float* w = (const float*)d_in[1];
  float* out = (float*)d_out;
  dim3 grid(B_SZ / BM, G_SZ);   // 4 m-tiles x 100 groups
  groupasl_gemm<<<grid, 256, 0, stream>>>(h, w, out);
}
